// LSTMOneHot_11759620457081
// MI455X (gfx1250) — compile-verified, matrix operand order checked by reading, not run on hardware
//
#include <hip/hip_runtime.h>

// ---------------------------------------------------------------------------
// 2-layer LSTM LM for MI455X (gfx1250, wave32, WMMA bf16->f32).
//   logits = LSTM2(embed(x)) @ fc_w^T + fc_b
// Design:
//   * weights pre-converted fp32 -> bf16 (prologue kernel)
//   * layer recurrence: 32 persistent workgroups, recurrent weight slice
//     (128 x 1024 bf16 = 256KB) pinned in LDS; h double-buffered in global
//     (bf16); c in registers; grid spin-barrier between the 512 steps.
//   * throughput GEMMs (xp1, logits): 128x128 block tile, 8 waves,
//     v_wmma_f32_16x16x32_bf16, double-buffered K-panels filled with
//     GLOBAL_LOAD_ASYNC_TO_LDS_B128 (ASYNCcnt) so global fetch overlaps WMMA.
// Workspace use: ~370 MB.
// ---------------------------------------------------------------------------

typedef __bf16 bf16_t;
typedef __attribute__((ext_vector_type(16))) __bf16 v16bf;
typedef __attribute__((ext_vector_type(8)))  float  v8f;

#define Bsz  32
#define Tlen 512
#define Hdim 1024
#define Vdim 4096
#define G4   (4 * Hdim)

__device__ __forceinline__ bf16_t f2bf(float f) {
  unsigned u = __float_as_uint(f);
  unsigned short s = (unsigned short)((u + 0x7FFFu + ((u >> 16) & 1u)) >> 16);
  bf16_t r;
  __builtin_memcpy(&r, &s, sizeof(r));
  return r;
}

// 32-byte WMMA fragment load as two 16B pieces (works for LDS & global, only
// needs 16B alignment).
__device__ __forceinline__ v16bf ld_frag(const bf16_t* p) {
  union { uint4 u[2]; v16bf v; } x;
  x.u[0] = *(const uint4*)(p);
  x.u[1] = *(const uint4*)(p + 8);
  return x.v;
}

__device__ __forceinline__ float sigmoidf_(float x) {
  return 1.0f / (1.0f + __expf(-x));
}

// Issue one 16B global->LDS async copy (GLOBAL_LOAD_ASYNC_TO_LDS_B128,
// tracked by ASYNCcnt). ldsAddr = allocation-relative LDS byte address
// (= low 32 bits of generic __shared__ pointer).
__device__ __forceinline__ void async_g2l_b128(const void* gptr, void* sptr) {
  unsigned lds = (unsigned)(unsigned long long)(uintptr_t)sptr;
  unsigned long long ga = (unsigned long long)(uintptr_t)gptr;
  asm volatile("global_load_async_to_lds_b128 %0, %1, off"
               :: "v"(lds), "v"(ga)
               : "memory");
}

__device__ __forceinline__ void wait_async0() {
  asm volatile("s_wait_asynccnt 0" ::: "memory");
}

// ---------------------------------------------------------------------------
// fp32 -> bf16 conversion (grid-stride)
// ---------------------------------------------------------------------------
__global__ void f32_to_bf16_kernel(const float* __restrict__ s,
                                   bf16_t* __restrict__ d, int n) {
  int i = blockIdx.x * blockDim.x + threadIdx.x;
  int stride = gridDim.x * blockDim.x;
  for (; i < n; i += stride) d[i] = f2bf(s[i]);
}

// ---------------------------------------------------------------------------
// recurrence init: h0 (f32) -> hbuf[0] (bf16), zero spin counter
// ---------------------------------------------------------------------------
__global__ void rec_init_kernel(const float* __restrict__ h0,
                                bf16_t* __restrict__ hbuf,
                                unsigned* __restrict__ cnt) {
  int i = blockIdx.x * blockDim.x + threadIdx.x;
  if (i < Bsz * Hdim) hbuf[i] = f2bf(h0[i]);
  if (i == 0) *cnt = 0u;
}

// ---------------------------------------------------------------------------
// LSTM recurrence (one layer). Grid = 32 blocks x 256 threads (8 waves).
// Block b owns h-columns [32b, 32b+32) => gate rows {g*H + 32b .. +32} for
// g in {i,f,g,o}; those 128 weight rows (x K=1024, bf16) live in LDS.
// Dynamic LDS: 128*1024*2 (weights) + 32*128*4 (gate staging) = 272KB.
// ---------------------------------------------------------------------------
__global__ void __launch_bounds__(256)
lstm_rec_kernel(const bf16_t* __restrict__ whh,   // [4H, H] bf16
                const float*  __restrict__ bhh,   // [4H]
                const float*  __restrict__ xp,    // [B,T,4H] f32, or nullptr
                const int*    __restrict__ tok,   // [B,T]    (layer 0)
                const float*  __restrict__ wih0,  // [4H, V]  (layer 0)
                const float*  __restrict__ bih0,  // [4H]     (layer 0)
                const float*  __restrict__ c_in,  // [B,H]
                bf16_t* __restrict__ hbuf,        // [2][B*H] bf16 (double buf)
                bf16_t* __restrict__ outh,        // [B,T,H] bf16
                float*  __restrict__ hT,          // [B,H] f32
                float*  __restrict__ cT,          // [B,H] f32
                unsigned* __restrict__ cnt) {
  extern __shared__ char smem[];
  bf16_t* wlds = (bf16_t*)smem;                                   // [128][1024]
  float*  glds = (float*)(smem + (size_t)128 * Hdim * sizeof(bf16_t)); // [32][128]

  const int tid  = threadIdx.x;
  const int lane = tid & 31;
  const int wave = tid >> 5;           // 0..7
  const int hc0  = blockIdx.x * 32;    // this block's h-column base

  // --- pin this block's 128 weight rows into LDS (row r -> global row
  //     (r>>5)*H + hc0 + (r&31)), 16B vectorized ---
  {
    const int vpr = Hdim / 8;  // uint4 per row
    for (int i = tid; i < 128 * vpr; i += 256) {
      int r = i / vpr, v = i - (i / vpr) * vpr;
      int gr = (r >> 5) * Hdim + hc0 + (r & 31);
      ((uint4*)wlds)[(size_t)r * vpr + v] =
          ((const uint4*)(whh + (size_t)gr * Hdim))[v];
    }
  }

  // --- cell state in registers: thread owns 4 (m, col) pairs for all T ---
  float c_reg[4];
  int pm[4], pc[4];
#pragma unroll
  for (int q = 0; q < 4; ++q) {
    int p = q * 256 + tid;          // 0..1023
    pm[q] = p >> 5;                 // batch row 0..31
    pc[q] = p & 31;                 // local h-col 0..31
    c_reg[q] = c_in[pm[q] * Hdim + hc0 + pc[q]];
  }
  __syncthreads();

  const int lr0  = wave * 16;          // wave's local gate-row tile
  const int rsel = lane & 15;
  const int ksel = (lane >> 4) << 4;

  for (int t = 0; t < Tlen; ++t) {
    const bf16_t* hsrc = hbuf + (size_t)(t & 1) * (Bsz * Hdim);
    bf16_t*       hdst = hbuf + (size_t)((t + 1) & 1) * (Bsz * Hdim);

    // ---- gates += h @ W_hh^T  (wave: C tiles [0..16)x[lr0..lr0+16) and
    //      [16..32)x[lr0..lr0+16), K=1024) ----
    v8f acc0 = {};
    v8f acc1 = {};
#pragma unroll 4
    for (int k0 = 0; k0 < Hdim; k0 += 32) {
      v16bf b  = ld_frag(wlds + (size_t)(lr0 + rsel) * Hdim + k0 + ksel);
      v16bf a0 = ld_frag(hsrc + (size_t)(rsel) * Hdim + k0 + ksel);
      v16bf a1 = ld_frag(hsrc + (size_t)(16 + rsel) * Hdim + k0 + ksel);
      acc0 = __builtin_amdgcn_wmma_f32_16x16x32_bf16(false, a0, false, b,
                                                     (short)0, acc0, false, false);
      acc1 = __builtin_amdgcn_wmma_f32_16x16x32_bf16(false, a1, false, b,
                                                     (short)0, acc1, false, false);
    }
    // stage accumulators in LDS: glds[m][local_gate_row]
    {
      int n  = lr0 + (lane & 15);
      int mb = (lane >> 4) << 3;
#pragma unroll
      for (int v = 0; v < 8; ++v) {
        glds[(mb + v) * 128 + n]        = acc0[v];
        glds[(16 + mb + v) * 128 + n]   = acc1[v];
      }
    }
    __syncthreads();

    // ---- elementwise LSTM cell for the 32x32 h-tile this block owns ----
#pragma unroll
    for (int q = 0; q < 4; ++q) {
      int m = pm[q], lc = pc[q];
      int hcol = hc0 + lc;
      float gi = glds[m * 128 + lc];
      float gf = glds[m * 128 + 32 + lc];
      float gg = glds[m * 128 + 64 + lc];
      float go = glds[m * 128 + 96 + lc];
      float xi, xf, xg, xo;
      if (xp != nullptr) {
        const float* xr = xp + ((size_t)(m * Tlen + t)) * G4;
        xi = xr[hcol];
        xf = xr[Hdim + hcol];
        xg = xr[2 * Hdim + hcol];
        xo = xr[3 * Hdim + hcol];
      } else {
        // layer 0: one-hot projection == gather of w_ih_0 column
        int v = tok[m * Tlen + t];
        xi = wih0[(size_t)(hcol) * Vdim + v]            + bih0[hcol];
        xf = wih0[(size_t)(Hdim + hcol) * Vdim + v]     + bih0[Hdim + hcol];
        xg = wih0[(size_t)(2 * Hdim + hcol) * Vdim + v] + bih0[2 * Hdim + hcol];
        xo = wih0[(size_t)(3 * Hdim + hcol) * Vdim + v] + bih0[3 * Hdim + hcol];
      }
      gi += xi + bhh[hcol];
      gf += xf + bhh[Hdim + hcol];
      gg += xg + bhh[2 * Hdim + hcol];
      go += xo + bhh[3 * Hdim + hcol];
      float si = sigmoidf_(gi);
      float sf = sigmoidf_(gf);
      float so = sigmoidf_(go);
      float c  = sf * c_reg[q] + si * tanhf(gg);
      c_reg[q] = c;
      float h  = so * tanhf(c);
      bf16_t hb = f2bf(h);
      hdst[m * Hdim + hcol] = hb;
      outh[((size_t)(m * Tlen + t)) * Hdim + hcol] = hb;
      if (t == Tlen - 1) {
        hT[m * Hdim + hcol] = h;
        cT[m * Hdim + hcol] = c;
      }
    }

    // ---- device-scope barrier between timesteps (32 resident blocks) ----
    __syncthreads();
    if (tid == 0) {
      __threadfence();
      atomicAdd(cnt, 1u);
      unsigned target = (unsigned)(t + 1) * gridDim.x;
      while (*(volatile unsigned*)cnt < target) {
        __builtin_amdgcn_s_sleep(1);
      }
      __threadfence();
    }
    __syncthreads();
  }
}

// ---------------------------------------------------------------------------
// C[M,N] f32 = A[M,K] bf16 (row-major) x W[N,K] bf16 (row-major, i.e. B^T)
//            + bias[N]
// Block tile 128x128, 256 threads = 8 waves (2 m-waves x 4 n-waves),
// wave tile 64x32 => 4x2 WMMA accumulators.
// K-panels (32 wide) double-buffered in LDS and filled with
// GLOBAL_LOAD_ASYNC_TO_LDS_B128: the next panel's copies are in flight while
// the current panel feeds 8 WMMAs per wave; one s_wait_asynccnt + one barrier per
// k-step.
// ---------------------------------------------------------------------------
#define GLDP 40  // padded LDS row length (elems) for a 32-wide K panel

__global__ void __launch_bounds__(256)
gemm_bf16_kernel(const bf16_t* __restrict__ A,
                 const bf16_t* __restrict__ W,
                 const float*  __restrict__ bias,
                 float* __restrict__ C,
                 int M, int N, int K) {
  __shared__ bf16_t aP[2][128 * GLDP];
  __shared__ bf16_t wP[2][128 * GLDP];

  const int tid  = threadIdx.x;
  const int lane = tid & 31;
  const int wave = tid >> 5;
  const int mw   = wave >> 2;   // 0..1
  const int nw   = wave & 3;    // 0..3
  const int bm   = blockIdx.y * 128;
  const int bn   = blockIdx.x * 128;

  v8f acc[4][2] = {};
  const int rsel = lane & 15;
  const int ksel = (lane >> 4) << 4;

  // async-stage one 32-wide K panel pair (A and W) into LDS buffer `buf`.
  // Panel = 128 rows x 32 bf16 = 512 x 16B chunks; 2 chunks/thread/matrix.
  auto stage = [&](int buf, int k0) {
#pragma unroll
    for (int c = tid; c < 512; c += 256) {
      int row  = c >> 2;        // 0..127
      int part = c & 3;         // 16B chunk within the 64B row
      async_g2l_b128(A + (size_t)(bm + row) * K + k0 + part * 8,
                     &aP[buf][row * GLDP + part * 8]);
      async_g2l_b128(W + (size_t)(bn + row) * K + k0 + part * 8,
                     &wP[buf][row * GLDP + part * 8]);
    }
  };

  stage(0, 0);

  for (int k0 = 0; k0 < K; k0 += 32) {
    const int cur = (k0 >> 5) & 1;
    wait_async0();          // this wave's async copies into `cur` have landed
    __syncthreads();        // ... and everyone else's too
    if (k0 + 32 < K) stage(cur ^ 1, k0 + 32);  // prefetch next panel async

    v16bf bfr[2];
#pragma unroll
    for (int j = 0; j < 2; ++j)
      bfr[j] = ld_frag(&wP[cur][(size_t)(nw * 32 + j * 16 + rsel) * GLDP + ksel]);
#pragma unroll
    for (int i = 0; i < 4; ++i) {
      v16bf a = ld_frag(&aP[cur][(size_t)(mw * 64 + i * 16 + rsel) * GLDP + ksel]);
#pragma unroll
      for (int j = 0; j < 2; ++j) {
        acc[i][j] = __builtin_amdgcn_wmma_f32_16x16x32_bf16(
            false, a, false, bfr[j], (short)0, acc[i][j], false, false);
      }
    }
  }

  // epilogue: bias + store f32
#pragma unroll
  for (int i = 0; i < 4; ++i) {
#pragma unroll
    for (int j = 0; j < 2; ++j) {
      int n  = bn + nw * 32 + j * 16 + (lane & 15);
      float bv = bias[n];
      int mb = bm + mw * 64 + i * 16 + ((lane >> 4) << 3);
#pragma unroll
      for (int v = 0; v < 8; ++v) {
        C[(size_t)(mb + v) * N + n] = acc[i][j][v] + bv;
      }
    }
  }
}

// ---------------------------------------------------------------------------
extern "C" void kernel_launch(void* const* d_in, const int* in_sizes, int n_in,
                              void* d_out, int out_size, void* d_ws, size_t ws_size,
                              hipStream_t stream) {
  (void)in_sizes; (void)n_in; (void)out_size; (void)ws_size;
  const int*   x      = (const int*)  d_in[0];
  const float* hidden = (const float*)d_in[1];   // [2,B,H]
  const float* memory = (const float*)d_in[2];   // [2,B,H]
  const float* w_ih_0 = (const float*)d_in[3];   // [4H,V]
  const float* w_hh_0 = (const float*)d_in[4];   // [4H,H]
  const float* b_ih_0 = (const float*)d_in[5];
  const float* b_hh_0 = (const float*)d_in[6];
  const float* w_ih_1 = (const float*)d_in[7];   // [4H,H]
  const float* w_hh_1 = (const float*)d_in[8];   // [4H,H]
  const float* b_ih_1 = (const float*)d_in[9];
  const float* b_hh_1 = (const float*)d_in[10];
  const float* fc_w   = (const float*)d_in[11];  // [V,H]
  const float* fc_b   = (const float*)d_in[12];
  float* out = (float*)d_out;

  // ---- workspace carve-out (~370 MB) ----
  char* ws = (char*)d_ws;
  size_t off = 0;
  auto take = [&](size_t bytes) -> char* {
    char* p = ws + off;
    off += (bytes + 255) & ~(size_t)255;
    return p;
  };
  const size_t WEL = (size_t)G4 * Hdim;  // 4M elements
  bf16_t* whh0b = (bf16_t*)take(WEL * 2);
  bf16_t* whh1b = (bf16_t*)take(WEL * 2);
  bf16_t* wih1b = (bf16_t*)take(WEL * 2);
  bf16_t* fcwb  = (bf16_t*)take((size_t)Vdim * Hdim * 2);
  float*  xpbuf = (float*) take((size_t)Bsz * Tlen * G4 * 4);
  bf16_t* out0b = (bf16_t*)take((size_t)Bsz * Tlen * Hdim * 2);
  bf16_t* out1b = (bf16_t*)take((size_t)Bsz * Tlen * Hdim * 2);
  bf16_t* hbuf  = (bf16_t*)take((size_t)2 * Bsz * Hdim * 2);
  unsigned* cnt = (unsigned*)take(256);

  const size_t LOGITS = (size_t)Bsz * Tlen * Vdim;
  float* hT0 = out + LOGITS;
  float* hT1 = out + LOGITS + (size_t)Bsz * Hdim;
  float* cT0 = out + LOGITS + (size_t)2 * Bsz * Hdim;
  float* cT1 = out + LOGITS + (size_t)3 * Bsz * Hdim;

  // ---- prologue: weights fp32 -> bf16 ----
  f32_to_bf16_kernel<<<2048, 256, 0, stream>>>(w_hh_0, whh0b, (int)WEL);
  f32_to_bf16_kernel<<<2048, 256, 0, stream>>>(w_hh_1, whh1b, (int)WEL);
  f32_to_bf16_kernel<<<2048, 256, 0, stream>>>(w_ih_1, wih1b, (int)WEL);
  f32_to_bf16_kernel<<<2048, 256, 0, stream>>>(fc_w,   fcwb,  Vdim * Hdim);

  const size_t recLds =
      (size_t)128 * Hdim * sizeof(bf16_t) + (size_t)Bsz * 128 * sizeof(float);

  // ---- layer 0 (fused one-hot gather) ----
  rec_init_kernel<<<(Bsz * Hdim + 255) / 256, 256, 0, stream>>>(hidden, hbuf, cnt);
  lstm_rec_kernel<<<Hdim / 32, 256, recLds, stream>>>(
      whh0b, b_hh_0, /*xp=*/nullptr, x, w_ih_0, b_ih_0,
      memory, hbuf, out0b, hT0, cT0, cnt);

  // ---- inter-layer projection: xp1 = out0 @ w_ih_1^T + b_ih_1 ----
  dim3 gg(G4 / 128, (Bsz * Tlen) / 128);
  gemm_bf16_kernel<<<gg, 256, 0, stream>>>(out0b, wih1b, b_ih_1, xpbuf,
                                           Bsz * Tlen, G4, Hdim);

  // ---- layer 1 ----
  rec_init_kernel<<<(Bsz * Hdim + 255) / 256, 256, 0, stream>>>(
      hidden + (size_t)Bsz * Hdim, hbuf, cnt);
  lstm_rec_kernel<<<Hdim / 32, 256, recLds, stream>>>(
      whh1b, b_hh_1, xpbuf, nullptr, nullptr, nullptr,
      memory + (size_t)Bsz * Hdim, hbuf, out1b, hT1, cT1, cnt);

  // ---- logits = out1 @ fc_w^T + fc_b ----
  gemm_bf16_kernel<<<gg, 256, 0, stream>>>(out1b, fcwb, fc_b, out,
                                           Bsz * Tlen, Vdim, Hdim);
}
